// lstm_encoder_24034636988673
// MI455X (gfx1250) — compile-verified, matrix operand order checked by reading, not run on hardware
//
#include <hip/hip_runtime.h>
#include <hip/hip_bf16.h>

// Problem constants (match reference).
#define Bb 256
#define Ll 128
#define Cc 256
#define Hh 512
#define Gg (4 * Hh)   // 2048

typedef __attribute__((ext_vector_type(8)))  __bf16 v8bf;
typedef __attribute__((ext_vector_type(16))) __bf16 v16bf;
typedef __attribute__((ext_vector_type(8)))  float  v8f;
typedef __attribute__((ext_vector_type(4)))  int    v4i;

#define AS1 __attribute__((address_space(1)))
#define AS3 __attribute__((address_space(3)))

#if __has_builtin(__builtin_amdgcn_global_load_async_to_lds_b128)
#define HAVE_ASYNC_LDS 1
#else
#define HAVE_ASYNC_LDS 0
#endif
#if __has_builtin(__builtin_amdgcn_s_wait_asynccnt)
#define WAIT_ASYNC(n) __builtin_amdgcn_s_wait_asynccnt(n)
#else
#define WAIT_ASYNC(n) asm volatile("s_wait_asynccnt %0" ::"n"(n))
#endif

// ---------------------------------------------------------------------------
// Prep kernels: bf16 conversion, weight transpose, x_score, state init
// ---------------------------------------------------------------------------

// x (B,L,C) f32 -> xb (L,B,C) bf16 (t-major so per-step A matrix is contiguous)
__global__ void cvt_x_kernel(const float* __restrict__ x, __bf16* __restrict__ xb) {
    int o = blockIdx.x * 256 + threadIdx.x;          // o indexes output (l,b,c)
    int c = o % Cc;
    int b = (o / Cc) % Bb;
    int l = o / (Cc * Bb);
    xb[o] = (__bf16)x[((size_t)b * Ll + l) * Cc + c];
}

// W (K,N) f32 row-major -> WT (N,K) bf16 row-major
__global__ void cvt_wT_kernel(const float* __restrict__ W, __bf16* __restrict__ WT,
                              int K, int N) {
    int o = blockIdx.x * 256 + threadIdx.x;
    if (o >= K * N) return;
    int k = o % K;
    int n = o / K;
    WT[o] = (__bf16)W[(size_t)k * N + n];
}

// x_score[b,c] = sum_l x[b,l,c] * Wa_x[l] + ba
__global__ void xscore_kernel(const float* __restrict__ x, const float* __restrict__ Wa,
                              const float* __restrict__ ba, float* __restrict__ xs) {
    int o = blockIdx.x * 256 + threadIdx.x;          // B*C
    int b = o / Cc, c = o % Cc;
    const float* wx = Wa + 2 * Hh;
    float s = ba[0];
    for (int l = 0; l < Ll; ++l)
        s += x[((size_t)b * Ll + l) * Cc + c] * wx[l];
    xs[o] = s;
}

__global__ void init_state_kernel(float* __restrict__ hf, float* __restrict__ cf,
                                  __bf16* __restrict__ hb) {
    int o = blockIdx.x * 256 + threadIdx.x;          // B*H
    hf[o] = 0.0f;
    cf[o] = 0.0f;
    hb[o] = (__bf16)0.0f;
}

// ---------------------------------------------------------------------------
// Per-step scalar kernels
// ---------------------------------------------------------------------------

// s_h[b] = h[b,:].Wa_h ; s_c[b] = C[b,:].Wa_c
__global__ void scores_kernel(const float* __restrict__ hf, const float* __restrict__ cf,
                              const float* __restrict__ Wa,
                              float* __restrict__ sh, float* __restrict__ sc) {
    __shared__ float r1[128];
    __shared__ float r2[128];
    int b = blockIdx.x, t = threadIdx.x;
    float a = 0.0f, d = 0.0f;
    for (int i = t; i < Hh; i += 128) {
        a += hf[b * Hh + i] * Wa[i];
        d += cf[b * Hh + i] * Wa[Hh + i];
    }
    r1[t] = a; r2[t] = d;
    __syncthreads();
    for (int s = 64; s > 0; s >>= 1) {
        if (t < s) { r1[t] += r1[t + s]; r2[t] += r2[t + s]; }
        __syncthreads();
    }
    if (t == 0) { sh[b] = r1[0]; sc[b] = r2[0]; }
}

// softmax over C of (x_score + s_h + s_c), x_tilde = alpha * x[:,t,:]
__global__ void softmax_kernel(const float* __restrict__ x, const float* __restrict__ xs,
                               const float* __restrict__ sh, const float* __restrict__ sc,
                               float* __restrict__ outX, int t) {
    __shared__ float red[256];
    int b = blockIdx.x, c = threadIdx.x;
    float s = xs[b * Cc + c] + sh[b] + sc[b];
    red[c] = s;
    __syncthreads();
    for (int st = 128; st > 0; st >>= 1) {
        if (c < st) red[c] = fmaxf(red[c], red[c + st]);
        __syncthreads();
    }
    float mx = red[0];
    __syncthreads();
    float e = __expf(s - mx);
    red[c] = e;
    __syncthreads();
    for (int st = 128; st > 0; st >>= 1) {
        if (c < st) red[c] += red[c + st];
        __syncthreads();
    }
    float inv = 1.0f / red[0];
    outX[((size_t)b * Ll + t) * Cc + c] = e * inv * x[((size_t)b * Ll + t) * Cc + c];
}

__device__ __forceinline__ float sigm(float v) { return 1.0f / (1.0f + __expf(-v)); }

// LSTM cell update from gates (B,4H): i | f | g | o
__global__ void update_kernel(const float* __restrict__ gates,
                              float* __restrict__ hf, float* __restrict__ cf,
                              __bf16* __restrict__ hb,
                              float* __restrict__ outH, float* __restrict__ outC, int t) {
    int o  = blockIdx.x * 256 + threadIdx.x;         // B*H
    int b  = o / Hh, hh = o % Hh;
    const float* g = gates + (size_t)b * Gg;
    float iv = g[hh];
    float fv = g[Hh + hh];
    float gv = g[2 * Hh + hh];
    float ov = g[3 * Hh + hh];
    float cn = sigm(fv) * cf[o] + sigm(iv) * tanhf(gv);
    float hn = sigm(ov) * tanhf(cn);
    cf[o] = cn;
    hf[o] = hn;
    hb[o] = (__bf16)hn;
    outH[((size_t)b * Ll + t) * Hh + hh] = hn;
    outC[((size_t)b * Ll + t) * Hh + hh] = cn;
}

// ---------------------------------------------------------------------------
// WMMA bf16 GEMM core with async-to-LDS double-buffered A-tile staging.
//
// Block = 128x128 output, 8 waves in a 4(M) x 2(N) grid; wave tile = 32x64
// (2x4 fragments of 16x16, f32 acc).  A (activations) is staged to LDS with
// GLOBAL_LOAD_ASYNC_TO_LDS_B128 per 64-wide K slice; B (pre-transposed
// weights, N-major) is read straight from global (L2-resident).
//
// ISA 16-bit fragment layout: lanes 0-15 -> row = lane, K in [k..k+7]U[k+16..k+23]
//                             lanes 16-31 -> row = lane-16, K shifted by +8.
// ---------------------------------------------------------------------------

#define KSTEP  64
#define LDSROW 72                       // 144B row pitch: conflict-free, 16B aligned
#define ATILE  (128 * LDSROW)           // one buffer, elements

__device__ __forceinline__ v16bf load_frag16(const __bf16* p) {
    v8bf lo = *(const v8bf*)p;          // K .. K+7   (16B aligned)
    v8bf hi = *(const v8bf*)(p + 16);   // K+16 .. K+23
    return __builtin_shufflevector(lo, hi, 0, 1, 2, 3, 4, 5, 6, 7,
                                           8, 9, 10, 11, 12, 13, 14, 15);
}

// Stage a 128x64 bf16 A tile (block rows) into LDS: 1024 16B chunks, 4/thread.
__device__ __forceinline__ void stage_A_tile(const __bf16* __restrict__ A, int lda,
                                             int k0, __bf16* dst, int tid) {
#pragma unroll
    for (int it = 0; it < 4; ++it) {
        int q  = tid + it * 256;
        int r  = q >> 3;                 // tile row 0..127
        int k8 = (q & 7) << 3;           // K sub-offset 0,8,..,56
        const __bf16* src = A + (size_t)r * lda + k0 + k8;
        __bf16*       d   = dst + r * LDSROW + k8;
#if HAVE_ASYNC_LDS
        const AS1 void* gsrc = (const AS1 void*)src;
        AS3 void*       ldst = (AS3 void*)d;
        __builtin_amdgcn_global_load_async_to_lds_b128(
            (AS1 v4i*)gsrc, (AS3 v4i*)ldst, 0, 0);
#else
        *(v8bf*)d = *(const v8bf*)src;   // sync fallback: global load + ds_store
#endif
    }
}

__device__ __forceinline__ void gemm_lds_2x4(v8f acc[2][4],
        const __bf16* __restrict__ A, int lda,           // block-row base, M x K
        const __bf16* __restrict__ Bt, int ldb, int K,   // block-col base, N x K
        __bf16* sA, int tid, int wm, int wn, int half, int l15) {
    const int nK = K / KSTEP;
    stage_A_tile(A, lda, 0, sA, tid);
    int buf = 0;
    for (int kb = 0; kb < nK; ++kb) {
        const int k0 = kb * KSTEP;
        if (kb + 1 < nK) {
            stage_A_tile(A, lda, k0 + KSTEP, sA + (buf ^ 1) * ATILE, tid);
            __builtin_prefetch(Bt + (size_t)(wn * 64 + l15) * ldb + k0 + KSTEP, 0, 1);
#if HAVE_ASYNC_LDS
            WAIT_ASYNC(4);               // current tile's 4 async loads retired
#endif
        } else {
#if HAVE_ASYNC_LDS
            WAIT_ASYNC(0);
#endif
        }
        __syncthreads();                 // tile visible to all 8 waves
        const __bf16* sbuf = sA + buf * ATILE;
#pragma unroll
        for (int kk = 0; kk < KSTEP; kk += 32) {
            const int koff = kk + half * 8;
            v16bf af[2];
#pragma unroll
            for (int i = 0; i < 2; ++i)
                af[i] = load_frag16(sbuf + (wm * 32 + i * 16 + l15) * LDSROW + koff);
#pragma unroll
            for (int j = 0; j < 4; ++j) {
                v16bf bfj = load_frag16(Bt + (size_t)(wn * 64 + j * 16 + l15) * ldb
                                        + k0 + koff);
#pragma unroll
                for (int i = 0; i < 2; ++i)
                    acc[i][j] = __builtin_amdgcn_wmma_f32_16x16x32_bf16(
                        false, af[i], false, bfj, (short)0, acc[i][j], false, false);
            }
        }
        __syncthreads();                 // done reading buf before it is restaged
        buf ^= 1;
    }
}

// mt = xt@Wmx + h@Wmh + bmx + bmh  ->  bf16 (B x H).  grid (B/128, H/128), 256 thr.
__global__ __launch_bounds__(256, 1)
void mt_gemm_kernel(const __bf16* __restrict__ Xt, const __bf16* __restrict__ WmxT,
                    const __bf16* __restrict__ Hb, const __bf16* __restrict__ WmhT,
                    const float* __restrict__ bmx, const float* __restrict__ bmh,
                    __bf16* __restrict__ Mb) {
    __shared__ __attribute__((aligned(16))) __bf16 sA[2 * ATILE];
    const int tid  = threadIdx.x;
    const int wave = tid >> 5, lane = tid & 31;
    const int half = lane >> 4, l15 = lane & 15;
    const int wm = wave >> 1, wn = wave & 1;             // 4 x 2 wave grid
    const int rowB = blockIdx.x * 128;
    const int colB = blockIdx.y * 128;
    v8f acc[2][4] = {};
    gemm_lds_2x4(acc, Xt + (size_t)rowB * Cc, Cc, WmxT + (size_t)colB * Cc, Cc, Cc,
                 sA, tid, wm, wn, half, l15);
    gemm_lds_2x4(acc, Hb + (size_t)rowB * Hh, Hh, WmhT + (size_t)colB * Hh, Hh, Hh,
                 sA, tid, wm, wn, half, l15);
#pragma unroll
    for (int i = 0; i < 2; ++i)
#pragma unroll
        for (int j = 0; j < 4; ++j)
#pragma unroll
            for (int r = 0; r < 8; ++r) {
                int gm = rowB + wm * 32 + i * 16 + r + 8 * half; // D: M = r + 8*half
                int gn = colB + wn * 64 + j * 16 + l15;          //    N = lane & 15
                float v = acc[i][j][r] + bmx[gn] + bmh[gn];
                Mb[(size_t)gm * Hh + gn] = (__bf16)v;
            }
}

// gates = xt@Wih + mt@Whm + bih + bhm -> f32 (B x 4H). grid (B/128, 4H/128), 256 thr.
__global__ __launch_bounds__(256, 1)
void gates_gemm_kernel(const __bf16* __restrict__ Xt, const __bf16* __restrict__ WihT,
                       const __bf16* __restrict__ Mb, const __bf16* __restrict__ WhmT,
                       const float* __restrict__ bih, const float* __restrict__ bhm,
                       float* __restrict__ gates) {
    __shared__ __attribute__((aligned(16))) __bf16 sA[2 * ATILE];
    const int tid  = threadIdx.x;
    const int wave = tid >> 5, lane = tid & 31;
    const int half = lane >> 4, l15 = lane & 15;
    const int wm = wave >> 1, wn = wave & 1;
    const int rowB = blockIdx.x * 128;
    const int colB = blockIdx.y * 128;
    v8f acc[2][4] = {};
    gemm_lds_2x4(acc, Xt + (size_t)rowB * Cc, Cc, WihT + (size_t)colB * Cc, Cc, Cc,
                 sA, tid, wm, wn, half, l15);
    gemm_lds_2x4(acc, Mb + (size_t)rowB * Hh, Hh, WhmT + (size_t)colB * Hh, Hh, Hh,
                 sA, tid, wm, wn, half, l15);
#pragma unroll
    for (int i = 0; i < 2; ++i)
#pragma unroll
        for (int j = 0; j < 4; ++j)
#pragma unroll
            for (int r = 0; r < 8; ++r) {
                int gm = rowB + wm * 32 + i * 16 + r + 8 * half;
                int gn = colB + wn * 64 + j * 16 + l15;
                gates[(size_t)gm * Gg + gn] = acc[i][j][r] + bih[gn] + bhm[gn];
            }
}

// ---------------------------------------------------------------------------
// Host launch
// ---------------------------------------------------------------------------

extern "C" void kernel_launch(void* const* d_in, const int* in_sizes, int n_in,
                              void* d_out, int out_size, void* d_ws, size_t ws_size,
                              hipStream_t stream) {
    const float* x   = (const float*)d_in[0];
    const float* Wih = (const float*)d_in[1];
    const float* bih = (const float*)d_in[2];
    const float* Wmx = (const float*)d_in[3];
    const float* bmx = (const float*)d_in[4];
    const float* Wmh = (const float*)d_in[5];
    const float* bmh = (const float*)d_in[6];
    const float* Whm = (const float*)d_in[7];
    const float* bhm = (const float*)d_in[8];
    const float* Wa  = (const float*)d_in[9];
    const float* ba  = (const float*)d_in[10];

    float* outH = (float*)d_out;                     // (B,L,H)
    float* outC = outH + (size_t)Bb * Ll * Hh;       // (B,L,H)
    float* outX = outC + (size_t)Bb * Ll * Hh;       // (B,L,C)

    // Workspace carve-up (256B aligned).
    char*  base = (char*)d_ws;
    size_t off  = 0;
    auto carve = [&](size_t bytes) -> char* {
        char* p = base + off;
        off = (off + bytes + 255) & ~(size_t)255;
        return p;
    };
    __bf16* xb    = (__bf16*)carve((size_t)Bb * Ll * Cc * 2);  // (L,B,C)
    __bf16* wihT  = (__bf16*)carve((size_t)Gg * Cc * 2);       // (4H,C)
    __bf16* wmxT  = (__bf16*)carve((size_t)Hh * Cc * 2);       // (H,C)
    __bf16* wmhT  = (__bf16*)carve((size_t)Hh * Hh * 2);       // (H,H)
    __bf16* whmT  = (__bf16*)carve((size_t)Gg * Hh * 2);       // (4H,H)
    __bf16* hb    = (__bf16*)carve((size_t)Bb * Hh * 2);       // bf16 h state
    __bf16* mb    = (__bf16*)carve((size_t)Bb * Hh * 2);       // bf16 mt
    float*  hf    = (float*)carve((size_t)Bb * Hh * 4);        // f32 h state
    float*  cf    = (float*)carve((size_t)Bb * Hh * 4);        // f32 C state
    float*  gates = (float*)carve((size_t)Bb * Gg * 4);        // f32 gates
    float*  xsc   = (float*)carve((size_t)Bb * Cc * 4);        // x_score
    float*  sh    = (float*)carve((size_t)Bb * 4);
    float*  sc    = (float*)carve((size_t)Bb * 4);
    (void)ws_size; (void)in_sizes; (void)n_in; (void)out_size;

    // ---- one-time prep (re-run every call for determinism) ----
    cvt_x_kernel<<<(Bb * Ll * Cc) / 256, 256, 0, stream>>>(x, xb);
    cvt_wT_kernel<<<((size_t)Cc * Gg + 255) / 256, 256, 0, stream>>>(Wih, wihT, Cc, Gg);
    cvt_wT_kernel<<<((size_t)Cc * Hh + 255) / 256, 256, 0, stream>>>(Wmx, wmxT, Cc, Hh);
    cvt_wT_kernel<<<((size_t)Hh * Hh + 255) / 256, 256, 0, stream>>>(Wmh, wmhT, Hh, Hh);
    cvt_wT_kernel<<<((size_t)Hh * Gg + 255) / 256, 256, 0, stream>>>(Whm, whmT, Hh, Gg);
    xscore_kernel<<<(Bb * Cc) / 256, 256, 0, stream>>>(x, Wa, ba, xsc);
    init_state_kernel<<<(Bb * Hh) / 256, 256, 0, stream>>>(hf, cf, hb);

    // ---- recurrence over timesteps (stream order enforces dependencies) ----
    for (int t = 0; t < Ll; ++t) {
        const __bf16* Xt = xb + (size_t)t * Bb * Cc;
        scores_kernel<<<Bb, 128, 0, stream>>>(hf, cf, Wa, sh, sc);
        softmax_kernel<<<Bb, 256, 0, stream>>>(x, xsc, sh, sc, outX, t);
        mt_gemm_kernel<<<dim3(Bb / 128, Hh / 128), 256, 0, stream>>>(
            Xt, wmxT, hb, wmhT, bmx, bmh, mb);
        gates_gemm_kernel<<<dim3(Bb / 128, Gg / 128), 256, 0, stream>>>(
            Xt, wihT, mb, whmT, bih, bhm, gates);
        update_kernel<<<(Bb * Hh) / 256, 256, 0, stream>>>(
            gates, hf, cf, hb, outH, outC, t);
    }
}
